// SpatialMambaFusionBlock_65197603553692
// MI455X (gfx1250) — matrix operand order repaired and checked by reading, not compiled, not run on hardware
//
#include <hip/hip_runtime.h>
#include <hip/hip_bf16.h>

typedef __bf16 bf16;
typedef __attribute__((ext_vector_type(16))) __bf16 v16bf;
typedef __attribute__((ext_vector_type(8)))  float  v8f;

#define BB      8
#define CC      128
#define HH      64
#define WW      64
#define HWSZ    4096
#define DIN     256
#define DS      64
#define HD      32
#define NHEAD   8
#define CONVCH  384
#define NTOT    648      // DIN + CONVCH + NHEAD
#define NPAD    704      // NTOT padded to 64-col tiles
#define NBR     5

// ---------------------------------------------------------------- helpers
__device__ __forceinline__ float siluf(float x) { return x / (1.f + __expf(-x)); }
__device__ __forceinline__ float geluf(float x) {
  float x3 = x * x * x;
  return 0.5f * x * (1.f + tanhf(0.7978845608028654f * (x + 0.044715f * x3)));
}
__device__ __forceinline__ float softplusf(float x) {
  return (x > 20.f) ? x : log1pf(__expf(x));
}

// scan-direction permutation (closed form, H=W=64):
// dir0 row-major identity; dir1 transpose; dir2 horizontal snake; dir3 vertical snake
__device__ __forceinline__ int perm_idx(int dir, int j) {
  int hi = j >> 6, lo = j & 63;
  switch (dir) {
    case 0:  return j;
    case 1:  return (lo << 6) | hi;
    case 2:  return (hi << 6) | ((hi & 1) ? (63 - lo) : lo);
    default: { int b = (hi & 1) ? (63 - lo) : lo; return (b << 6) | hi; }
  }
}

__device__ __forceinline__ v8f zero8() {
  v8f r;
#pragma unroll
  for (int i = 0; i < 8; ++i) r[i] = 0.f;
  return r;
}

// A fragment, 16x32 bf16 (row-major source, contiguous K per row).
// lane<16: row=lane, elems 0..7 = K k0..k0+7, elems 8..15 = K k0+16..k0+23
// lane>=16: row=lane-16, elems 0..7 = K k0+8..15, elems 8..15 = K k0+24..31
__device__ __forceinline__ v16bf load_a_frag(const bf16* __restrict__ rowptr, int k0, int lane) {
  int kbase = k0 + ((lane & 16) ? 8 : 0);
  v16bf r;
  ((uint4*)&r)[0] = *(const uint4*)(rowptr + kbase);
  ((uint4*)&r)[1] = *(const uint4*)(rowptr + kbase + 16);
  return r;
}

// B fragment, 32x16 bf16 from transposed weights Bt[N][K] (row-major, ldb=K).
// lane<16: N=n0+lane, K=k0..k0+7,k0+16..k0+23 ; lane>=16: N=n0+lane-16, K=k0+8..k0+15,k0+24..k0+31
__device__ __forceinline__ v16bf load_b_frag(const bf16* __restrict__ Bt, int ldb,
                                             int n0, int k0, int lane) {
  const bf16* p = Bt + (size_t)(n0 + (lane & 15)) * ldb + k0 + ((lane & 16) ? 8 : 0);
  v16bf r;
  ((uint4*)&r)[0] = *(const uint4*)(p);
  ((uint4*)&r)[1] = *(const uint4*)(p + 16);
  return r;
}

#define WMMA_BF16(a, b, c) \
  __builtin_amdgcn_wmma_f32_16x16x32_bf16(false, (a), false, (b), (short)0, (c), false, false)

// ---------------------------------------------------------------- weight prep
// bf16-convert + transpose weights to [N][K] layout (WMMA B side).
__global__ void prep_w_kernel(const float* __restrict__ W_in, const float* __restrict__ W_out,
                              const float* __restrict__ proj_w,
                              bf16* __restrict__ Wt_in, bf16* __restrict__ Wt_out,
                              bf16* __restrict__ Wt_proj) {
  int i = blockIdx.x * blockDim.x + threadIdx.x;
  const int tot_in = 4 * NPAD * CC;
  const int tot_out = 4 * CC * DIN;
  const int tot_proj = CC * 640;
  if (i < tot_in) {
    int k = i % CC; int n = (i / CC) % NPAD; int d = i / (CC * NPAD);
    float v = (n < NTOT) ? W_in[((size_t)d * CC + k) * NTOT + n] : 0.f;
    Wt_in[i] = (bf16)v;
  } else if (i < tot_in + tot_out) {
    int j = i - tot_in;
    int k = j % DIN; int n = (j / DIN) % CC; int d = j / (DIN * CC);
    Wt_out[j] = (bf16)W_out[((size_t)d * DIN + k) * CC + n];
  } else if (i < tot_in + tot_out + tot_proj) {
    int j = i - tot_in - tot_out;
    Wt_proj[j] = (bf16)proj_w[j];   // proj_w (C_out, 640) is already [N][K]
  }
}

// ---------------------------------------------------------------- positional embedding
__global__ void posproj_kernel(const float* __restrict__ pw, float* __restrict__ posproj) {
  int hw = blockIdx.x;
  int c = threadIdx.x;              // 128
  int h = hw >> 6, w = hw & 63;
  __shared__ float pos[CC];
  float val;
  if (c < 64) {
    float ang = (float)h / __powf(10000.f, (float)(c & ~1) / 64.f);
    val = (c & 1) ? __cosf(ang) : __sinf(ang);
  } else {
    int cc = c - 64;
    float ang = (float)w / __powf(10000.f, (float)(cc & ~1) / 64.f);
    val = (cc & 1) ? __cosf(ang) : __sinf(ang);
  }
  pos[c] = val;
  __syncthreads();
  float s = 0.f;
#pragma unroll 8
  for (int k = 0; k < CC; ++k) s += pos[k] * pw[c * CC + k];
  posproj[(size_t)c * HWSZ + hw] = s;
}

__global__ void xp_kernel(const float* __restrict__ x, const float* __restrict__ posproj,
                          float* __restrict__ xp, bf16* __restrict__ seqA) {
  int hw = blockIdx.x, b = blockIdx.y, c = threadIdx.x;
  size_t xi = ((size_t)b * CC + c) * HWSZ + hw;
  float v = x[xi] + posproj[(size_t)c * HWSZ + hw];
  xp[xi] = v;
  seqA[((size_t)b * HWSZ + hw) * CC + c] = (bf16)v;   // (B, L, C) bf16 for WMMA A
}

// ---------------------------------------------------------------- local branch (dw 3x3 + gelu)
__global__ void local_kernel(const float* __restrict__ xp, const float* __restrict__ lw,
                             float* __restrict__ stacked, bf16* __restrict__ concatA) {
  int hw = blockIdx.x, b = blockIdx.y, c = threadIdx.x;
  int h = hw >> 6, w = hw & 63;
  const float* xc = xp + ((size_t)b * CC + c) * HWSZ;
  float s = 0.f;
#pragma unroll
  for (int dh = -1; dh <= 1; ++dh)
#pragma unroll
    for (int dw = -1; dw <= 1; ++dw) {
      int hh = h + dh, ww = w + dw;
      if (hh >= 0 && hh < HH && ww >= 0 && ww < WW)
        s += lw[c * 9 + (dh + 1) * 3 + (dw + 1)] * xc[hh * WW + ww];
    }
  s = geluf(s);
  stacked[(((size_t)b * NBR + 0) * CC + c) * HWSZ + hw] = s;
  concatA[((size_t)b * HWSZ + hw) * (NBR * CC) + c] = (bf16)s;
}

// ---------------------------------------------------------------- WMMA GEMM: zxbcdt = perm(u) @ W_in
// block: 128 thr (4 waves), wave -> 16 rows x 64 cols ; grid (64, 11, B)
__global__ void gemm_in_kernel(const bf16* __restrict__ seqA, const bf16* __restrict__ Wt_in,
                               float* __restrict__ out, int dir) {
  int wave = threadIdx.x >> 5, lane = threadIdx.x & 31;
  int b = blockIdx.z;
  int m0 = blockIdx.x * 64 + wave * 16;
  int n0 = blockIdx.y * 64;
  int srow = perm_idx(dir, m0 + (lane & 15));
  const bf16* arow = seqA + ((size_t)b * HWSZ + srow) * CC;
  const bf16* wt = Wt_in + (size_t)dir * NPAD * CC;
  v8f acc[4];
#pragma unroll
  for (int t = 0; t < 4; ++t) acc[t] = zero8();
#pragma unroll
  for (int k0 = 0; k0 < CC; k0 += 32) {
    v16bf a = load_a_frag(arow, k0, lane);
#pragma unroll
    for (int t = 0; t < 4; ++t) {
      v16bf bf = load_b_frag(wt, CC, n0 + t * 16, k0, lane);
      acc[t] = WMMA_BF16(a, bf, acc[t]);
    }
  }
  float* obase = out + (size_t)b * HWSZ * NTOT;
  int rbase = (lane & 16) ? 8 : 0;
#pragma unroll
  for (int t = 0; t < 4; ++t) {
    int cidx = n0 + t * 16 + (lane & 15);
    if (cidx < NTOT) {
#pragma unroll
      for (int r = 0; r < 8; ++r)
        obase[(size_t)(m0 + rbase + r) * NTOT + cidx] = acc[t][r];
    }
  }
}

// ---------------------------------------------------------------- causal dw conv1d + silu + dt/dA
__global__ void mid_kernel(const float* __restrict__ zx, const float* __restrict__ conv_w,
                           const float* __restrict__ A_log, const float* __restrict__ dt_bias,
                           float* __restrict__ convX, float* __restrict__ dtv,
                           float* __restrict__ dAv, int dir) {
  int l = blockIdx.x, b = blockIdx.y;
  const float* cw = conv_w + dir * 4 * CONVCH;
  float* orow = convX + ((size_t)b * HWSZ + l) * CONVCH;
  for (int ch = threadIdx.x; ch < CONVCH; ch += blockDim.x) {
    float s = 0.f;
#pragma unroll
    for (int t = 0; t < 4; ++t) {
      int ls = l - 3 + t;
      if (ls >= 0)
        s += cw[t * CONVCH + ch] * zx[((size_t)b * HWSZ + ls) * NTOT + DIN + ch];
    }
    orow[ch] = siluf(s);
  }
  if (threadIdx.x < NHEAD) {
    int hd = threadIdx.x;
    float dtr = zx[((size_t)b * HWSZ + l) * NTOT + DIN + CONVCH + hd] + dt_bias[dir * NHEAD + hd];
    float dt = softplusf(dtr);
    dtv[((size_t)b * HWSZ + l) * NHEAD + hd] = dt;
    dAv[((size_t)b * HWSZ + l) * NHEAD + hd] = __expf(-__expf(A_log[dir * NHEAD + hd]) * dt);
  }
}

// ---------------------------------------------------------------- selective scan
// grid (NHEAD, B), block 256: thread owns (d = tid/8, 8 n-values); state in registers.
__global__ void scan_kernel(const float* __restrict__ convX, const float* __restrict__ dtv,
                            const float* __restrict__ dAv, float* __restrict__ yscan) {
  int hd = blockIdx.x, b = blockIdx.y;
  int d = threadIdx.x >> 3, g = threadIdx.x & 7, n0 = g * 8;
  float hs[8];
#pragma unroll
  for (int j = 0; j < 8; ++j) hs[j] = 0.f;
  for (int l = 0; l < HWSZ; ++l) {
    const float* row = convX + ((size_t)b * HWSZ + l) * CONVCH;
    __builtin_prefetch(row + CONVCH, 0, 0);           // next timestep -> global_prefetch
    float dA = dAv[((size_t)b * HWSZ + l) * NHEAD + hd];
    float dt = dtv[((size_t)b * HWSZ + l) * NHEAD + hd];
    float coef = dt * row[hd * HD + d];
    float part = 0.f;
#pragma unroll
    for (int j = 0; j < 8; ++j) {
      hs[j] = dA * hs[j] + coef * row[DIN + n0 + j];          // B_t
      part += hs[j] * row[DIN + DS + n0 + j];                 // C_t
    }
    part += __shfl_xor(part, 1);
    part += __shfl_xor(part, 2);
    part += __shfl_xor(part, 4);
    if (g == 0) yscan[((size_t)b * HWSZ + l) * DIN + hd * HD + d] = part;
  }
}

// ---------------------------------------------------------------- gate + RMSNorm -> bf16 A
__global__ void gate_kernel(const float* __restrict__ yscan, const float* __restrict__ convX,
                            const float* __restrict__ zx, const float* __restrict__ Dp,
                            const float* __restrict__ norm_w, bf16* __restrict__ gatedA, int dir) {
  int l = blockIdx.x, b = blockIdx.y, c = threadIdx.x;   // 256 threads
  size_t r = (size_t)b * HWSZ + l;
  float y = yscan[r * DIN + c] + Dp[dir * NHEAD + (c >> 5)] * convX[r * CONVCH + c];
  float z = zx[r * NTOT + c];
  y *= siluf(z);
  __shared__ float red[256];
  red[c] = y * y;
  __syncthreads();
  for (int s = 128; s > 0; s >>= 1) {
    if (c < s) red[c] += red[c + s];
    __syncthreads();
  }
  float inv = rsqrtf(red[0] / (float)DIN + 1e-5f);
  gatedA[r * DIN + c] = (bf16)(y * inv * norm_w[dir * DIN + c]);
}

// ---------------------------------------------------------------- WMMA GEMM: y @ W_out, scatter by perm
__global__ void gemm_out_kernel(const bf16* __restrict__ gatedA, const bf16* __restrict__ Wt_out,
                                float* __restrict__ stacked, bf16* __restrict__ concatA, int dir) {
  int wave = threadIdx.x >> 5, lane = threadIdx.x & 31;
  int b = blockIdx.z;
  int m0 = blockIdx.x * 64 + wave * 16;
  int n0 = blockIdx.y * 64;
  const bf16* arow = gatedA + ((size_t)b * HWSZ + m0 + (lane & 15)) * DIN;
  const bf16* wt = Wt_out + (size_t)dir * CC * DIN;
  v8f acc[4];
#pragma unroll
  for (int t = 0; t < 4; ++t) acc[t] = zero8();
#pragma unroll
  for (int k0 = 0; k0 < DIN; k0 += 32) {
    v16bf a = load_a_frag(arow, k0, lane);
#pragma unroll
    for (int t = 0; t < 4; ++t) {
      v16bf bf = load_b_frag(wt, DIN, n0 + t * 16, k0, lane);
      acc[t] = WMMA_BF16(a, bf, acc[t]);
    }
  }
  int rbase = (lane & 16) ? 8 : 0;
#pragma unroll
  for (int r = 0; r < 8; ++r) {
    int l = m0 + rbase + r;
    int hw = perm_idx(dir, l);   // inverse-perm scatter: out[PERM[l]] = y[l]
#pragma unroll
    for (int t = 0; t < 4; ++t) {
      int c = n0 + t * 16 + (lane & 15);
      float v = acc[t][r];
      stacked[(((size_t)b * NBR + 1 + dir) * CC + c) * HWSZ + hw] = v;
      concatA[((size_t)b * HWSZ + hw) * (NBR * CC) + (1 + dir) * CC + c] = (bf16)v;
    }
  }
}

// ---------------------------------------------------------------- SE descriptor + gate
__global__ void desc_kernel(const float* __restrict__ stacked, float* __restrict__ desc) {
  int idx = blockIdx.x;                       // b*5 + br
  const float* p = stacked + (size_t)idx * CC * HWSZ;
  float s = 0.f;
  for (size_t i = threadIdx.x; i < (size_t)CC * HWSZ; i += blockDim.x) s += p[i];
  __shared__ float red[256];
  red[threadIdx.x] = s;
  __syncthreads();
  for (int st = 128; st > 0; st >>= 1) {
    if (threadIdx.x < st) red[threadIdx.x] += red[threadIdx.x + st];
    __syncthreads();
  }
  if (threadIdx.x == 0) desc[idx] = red[0] / (float)((size_t)CC * HWSZ);
}

__global__ void se_kernel(const float* __restrict__ desc, const float* __restrict__ w1,
                          const float* __restrict__ b1, const float* __restrict__ w2,
                          const float* __restrict__ b2, float* __restrict__ wts) {
  int t = threadIdx.x;
  if (t >= BB * NBR) return;
  int b = t / NBR, br = t % NBR;
  float h0 = b1[0], h1 = b1[1];
#pragma unroll
  for (int i = 0; i < NBR; ++i) {
    float d = desc[b * NBR + i];
    h0 += d * w1[i * 2 + 0];
    h1 += d * w1[i * 2 + 1];
  }
  h0 = geluf(h0); h1 = geluf(h1);
  float o = b2[br] + h0 * w2[0 * NBR + br] + h1 * w2[1 * NBR + br];
  wts[t] = 1.f / (1.f + __expf(-o));
}

// ---------------------------------------------------------------- WMMA GEMM: proj = concat @ proj_w^T
__global__ void gemm_proj_kernel(const bf16* __restrict__ concatA, const bf16* __restrict__ Wt_proj,
                                 float* __restrict__ proj_ws) {
  int wave = threadIdx.x >> 5, lane = threadIdx.x & 31;
  int b = blockIdx.z;
  int m0 = blockIdx.x * 64 + wave * 16;
  int n0 = blockIdx.y * 64;
  const bf16* arow = concatA + ((size_t)b * HWSZ + m0 + (lane & 15)) * (NBR * CC);
  v8f acc[4];
#pragma unroll
  for (int t = 0; t < 4; ++t) acc[t] = zero8();
  for (int k0 = 0; k0 < NBR * CC; k0 += 32) {
    v16bf a = load_a_frag(arow, k0, lane);
#pragma unroll
    for (int t = 0; t < 4; ++t) {
      v16bf bf = load_b_frag(Wt_proj, NBR * CC, n0 + t * 16, k0, lane);
      acc[t] = WMMA_BF16(a, bf, acc[t]);
    }
  }
  int rbase = (lane & 16) ? 8 : 0;
#pragma unroll
  for (int t = 0; t < 4; ++t) {
    int c = n0 + t * 16 + (lane & 15);
#pragma unroll
    for (int r = 0; r < 8; ++r)
      proj_ws[((size_t)b * CC + c) * HWSZ + (m0 + rbase + r)] = acc[t][r];
  }
}

// ---------------------------------------------------------------- GroupNorm statistics
__global__ void gn_kernel(const float* __restrict__ proj_ws, float* __restrict__ gnstat) {
  int g = blockIdx.x, b = blockIdx.y;
  const float* p = proj_ws + ((size_t)b * CC + g * 16) * HWSZ;
  const int n = 16 * HWSZ;
  float s = 0.f, s2 = 0.f;
  for (int i = threadIdx.x; i < n; i += blockDim.x) {
    float v = p[i];
    s += v; s2 += v * v;
  }
  __shared__ float r1[256], r2[256];
  r1[threadIdx.x] = s; r2[threadIdx.x] = s2;
  __syncthreads();
  for (int st = 128; st > 0; st >>= 1) {
    if (threadIdx.x < st) { r1[threadIdx.x] += r1[threadIdx.x + st]; r2[threadIdx.x] += r2[threadIdx.x + st]; }
    __syncthreads();
  }
  if (threadIdx.x == 0) {
    float mean = r1[0] / (float)n;
    gnstat[(b * 8 + g) * 2 + 0] = mean;
    gnstat[(b * 8 + g) * 2 + 1] = r2[0] / (float)n - mean * mean;
  }
}

// ---------------------------------------------------------------- fused epilogue
__global__ void final_kernel(const float* __restrict__ x, const float* __restrict__ stacked,
                             const float* __restrict__ wts, const float* __restrict__ proj_ws,
                             const float* __restrict__ gnstat, const float* __restrict__ gamma,
                             const float* __restrict__ beta, float* __restrict__ out) {
  int hw = blockIdx.x, b = blockIdx.y, c = threadIdx.x;
  float fused = 0.f;
#pragma unroll
  for (int br = 0; br < NBR; ++br)
    fused += wts[b * NBR + br] * stacked[(((size_t)b * NBR + br) * CC + c) * HWSZ + hw];
  float p = proj_ws[((size_t)b * CC + c) * HWSZ + hw];
  int g = c >> 4;
  float mean = gnstat[(b * 8 + g) * 2 + 0];
  float var  = gnstat[(b * 8 + g) * 2 + 1];
  p = (p - mean) * rsqrtf(var + 1e-5f) * gamma[c] + beta[c];
  p = geluf(p);
  size_t o = ((size_t)b * CC + c) * HWSZ + hw;
  out[o] = x[o] + fused + p;
}

// ---------------------------------------------------------------- host launch
extern "C" void kernel_launch(void* const* d_in, const int* in_sizes, int n_in,
                              void* d_out, int out_size, void* d_ws, size_t ws_size,
                              hipStream_t stream) {
  (void)in_sizes; (void)n_in; (void)out_size;
  const float* x          = (const float*)d_in[0];
  const float* pos_proj_w = (const float*)d_in[1];
  const float* local_w    = (const float*)d_in[2];
  const float* W_in       = (const float*)d_in[3];
  const float* conv_w     = (const float*)d_in[4];
  const float* A_log      = (const float*)d_in[5];
  const float* dt_bias    = (const float*)d_in[6];
  const float* Dp         = (const float*)d_in[7];
  const float* norm_w     = (const float*)d_in[8];
  const float* W_out      = (const float*)d_in[9];
  const float* se_w1      = (const float*)d_in[10];
  const float* se_b1      = (const float*)d_in[11];
  const float* se_w2      = (const float*)d_in[12];
  const float* se_b2      = (const float*)d_in[13];
  const float* proj_w     = (const float*)d_in[14];
  const float* gn_gamma   = (const float*)d_in[15];
  const float* gn_beta    = (const float*)d_in[16];
  float* out = (float*)d_out;

  char* ws = (char*)d_ws;
  size_t off = 0;
  auto alloc = [&](size_t bytes) -> void* {
    off = (off + 255) & ~(size_t)255;
    void* p = ws + off;
    off += bytes;
    return p;
  };
  float* posproj = (float*)alloc(sizeof(float) * CC * HWSZ);
  float* xp      = (float*)alloc(sizeof(float) * BB * CC * HWSZ);
  bf16*  seqA    = (bf16*) alloc(sizeof(bf16)  * BB * HWSZ * CC);
  bf16*  Wt_in   = (bf16*) alloc(sizeof(bf16)  * 4 * NPAD * CC);
  bf16*  Wt_out  = (bf16*) alloc(sizeof(bf16)  * 4 * CC * DIN);
  bf16*  Wt_proj = (bf16*) alloc(sizeof(bf16)  * CC * NBR * CC);
  float* zx      = (float*)alloc(sizeof(float) * BB * HWSZ * NTOT);
  float* convX   = (float*)alloc(sizeof(float) * BB * HWSZ * CONVCH);
  float* dtv     = (float*)alloc(sizeof(float) * BB * HWSZ * NHEAD);
  float* dAv     = (float*)alloc(sizeof(float) * BB * HWSZ * NHEAD);
  float* yscan   = (float*)alloc(sizeof(float) * BB * HWSZ * DIN);
  bf16*  gatedA  = (bf16*) alloc(sizeof(bf16)  * BB * HWSZ * DIN);
  float* stacked = (float*)alloc(sizeof(float) * BB * NBR * CC * HWSZ);
  bf16*  concatA = (bf16*) alloc(sizeof(bf16)  * BB * HWSZ * NBR * CC);
  float* proj_ws = (float*)alloc(sizeof(float) * BB * CC * HWSZ);
  float* desc    = (float*)alloc(sizeof(float) * BB * NBR);
  float* wts     = (float*)alloc(sizeof(float) * BB * NBR);
  float* gnstat  = (float*)alloc(sizeof(float) * BB * 8 * 2);
  if (off > ws_size) return;   // insufficient workspace

  prep_w_kernel<<<2240, 256, 0, stream>>>(W_in, W_out, proj_w, Wt_in, Wt_out, Wt_proj);
  posproj_kernel<<<HWSZ, CC, 0, stream>>>(pos_proj_w, posproj);
  xp_kernel<<<dim3(HWSZ, BB), CC, 0, stream>>>(x, posproj, xp, seqA);
  local_kernel<<<dim3(HWSZ, BB), CC, 0, stream>>>(xp, local_w, stacked, concatA);

  for (int dir = 0; dir < 4; ++dir) {
    gemm_in_kernel<<<dim3(64, 11, BB), 128, 0, stream>>>(seqA, Wt_in, zx, dir);
    mid_kernel<<<dim3(HWSZ, BB), 128, 0, stream>>>(zx, conv_w, A_log, dt_bias, convX, dtv, dAv, dir);
    scan_kernel<<<dim3(NHEAD, BB), 256, 0, stream>>>(convX, dtv, dAv, yscan);
    gate_kernel<<<dim3(HWSZ, BB), 256, 0, stream>>>(yscan, convX, zx, Dp, norm_w, gatedA, dir);
    gemm_out_kernel<<<dim3(64, 2, BB), 128, 0, stream>>>(gatedA, Wt_out, stacked, concatA, dir);
  }

  desc_kernel<<<BB * NBR, 256, 0, stream>>>(stacked, desc);
  se_kernel<<<1, 64, 0, stream>>>(desc, se_w1, se_b1, se_w2, se_b2, wts);
  gemm_proj_kernel<<<dim3(64, 2, BB), 128, 0, stream>>>(concatA, Wt_proj, proj_ws);
  gn_kernel<<<dim3(8, BB), 256, 0, stream>>>(proj_ws, gnstat);
  final_kernel<<<dim3(HWSZ, BB), CC, 0, stream>>>(x, stacked, wts, proj_ws, gnstat,
                                                  gn_gamma, gn_beta, out);
}
